// StructformerPoincare_53858889892284
// MI455X (gfx1250) — matrix operand order repaired and checked by reading, not compiled, not run on hardware
//
#include <hip/hip_runtime.h>
#include <math.h>

// ---------------- model constants ----------------
#define VSZ 32000
#define DM  768
#define NH  12
#define NLAY 12
#define BB  2
#define TT  1024
#define HD  64            // DM / NH
#define MM  (BB*TT)       // 2048 rows of activations
#define LN_EPS 1e-6f
#define INV_SQRT_HD 0.125f
#define INV_SQRT_D  0.03608439182435161f   // 1/sqrt(768)
#define ALPHA_STRUCT 0.1f
#define DMAXC 4.0f

typedef __attribute__((ext_vector_type(16))) __bf16 v16bf;
typedef __attribute__((ext_vector_type(8)))  __bf16 v8bf;
typedef __attribute__((ext_vector_type(4)))  __bf16 v4bf;
typedef __attribute__((ext_vector_type(8)))  float  v8f;

// fp32 -> bf16 round-to-nearest-even
__device__ __forceinline__ __bf16 f2bf(float f) {
  union { float f; unsigned u; } a; a.f = f;
  unsigned u = a.u;
  u += 0x7FFFu + ((u >> 16) & 1u);
  union { unsigned short s; __bf16 b; } c; c.s = (unsigned short)(u >> 16);
  return c.b;
}

// ---------------- block reductions (blockDim.x == 256) ----------------
__device__ __forceinline__ float blk_sum(float v, float* sh) {
  int t = threadIdx.x;
  sh[t] = v; __syncthreads();
  for (int s = 128; s > 0; s >>= 1) { if (t < s) sh[t] += sh[t + s]; __syncthreads(); }
  float r = sh[0]; __syncthreads();
  return r;
}
__device__ __forceinline__ float blk_max(float v, float* sh) {
  int t = threadIdx.x;
  sh[t] = v; __syncthreads();
  for (int s = 128; s > 0; s >>= 1) { if (t < s) sh[t] = fmaxf(sh[t], sh[t + s]); __syncthreads(); }
  float r = sh[0]; __syncthreads();
  return r;
}

// ---------------- generic bf16 WMMA GEMM ----------------
// C[z] = A[z] (M x K, row stride lda) * B[z] (K x N, strides sbk/sbn) + bias
// block = 128 threads = 4 waves; block tile 32M x 256N; wave tile 32M x 64N
// (2 M-frags x 4 N-frags = 8 WMMAs per 32-deep K step).
// Requires: M % 32 == 0, K % 32 == 0, N % 64 == 0, and (sbn==1 || sbk==1).
// __launch_bounds__(128, 1): keep full VGPR budget -> no accumulator spills.
#define LBS 40   // LDS B row stride in bf16 (32 + 8 pad); 80B rows -> conflict-free
__global__ __launch_bounds__(128, 1)
void wmma_gemm(const float* __restrict__ A, long lda, long sAz,
               const float* __restrict__ B, long sbk, long sbn, long sBz,
               float* __restrict__ C, long ldc, long sCz,
               const float* __restrict__ bias,
               int M, int N, int K) {
  __shared__ __attribute__((aligned(16))) __bf16 lA[32 * 32];    // [m][k]
  __shared__ __attribute__((aligned(16))) __bf16 lB[256 * LBS];  // [n][k] transposed
  const int wave = threadIdx.x >> 5;
  const int lane = threadIdx.x & 31;
  const int m0 = blockIdx.x * 32;
  const int n0blk = blockIdx.y * 256;
  const int n0 = n0blk + wave * 64;
  const float* Ab = A + (long)blockIdx.z * sAz;
  const float* Bb = B + (long)blockIdx.z * sBz;
  float* Cb = C + (long)blockIdx.z * sCz;

  v8f acc[2][4] = {};
  const bool active = (n0 < N);           // wave-uniform -> EXEC stays all-ones
  const int am = lane & 15;
  const int kh = lane >> 4;

  for (int k0 = 0; k0 < K; k0 += 32) {
    // ---- stage A tile 32x32 (256 float4 / 128 threads) ----
#pragma unroll
    for (int i = 0; i < 2; ++i) {
      int e = threadIdx.x + i * 128;
      int r = e >> 3, c4 = (e & 7) * 4;
      float4 v = *(const float4*)(Ab + (long)(m0 + r) * lda + (k0 + c4));
      v4bf pk = {f2bf(v.x), f2bf(v.y), f2bf(v.z), f2bf(v.w)};
      *(v4bf*)&lA[r * 32 + c4] = pk;
    }
    // ---- stage B tile 32K x 256N transposed into lB[n][k] ----
    if (sbn == 1) {
      // rows contiguous in n: float4 along n, 4-way u16 scatter (transpose on store)
#pragma unroll 4
      for (int i = 0; i < 16; ++i) {
        int e = threadIdx.x + i * 128;          // 2048 float4s
        int k = e >> 6, n4 = (e & 63) * 4;
        int ng = n0blk + n4;
        float4 v = {0.f, 0.f, 0.f, 0.f};
        if (ng < N) v = *(const float4*)(Bb + (long)(k0 + k) * sbk + ng);
        lB[(n4 + 0) * LBS + k] = f2bf(v.x);
        lB[(n4 + 1) * LBS + k] = f2bf(v.y);
        lB[(n4 + 2) * LBS + k] = f2bf(v.z);
        lB[(n4 + 3) * LBS + k] = f2bf(v.w);
      }
    } else {
      // sbk==1 (transposed operand): columns contiguous in k -> b64 LDS stores
#pragma unroll 4
      for (int i = 0; i < 16; ++i) {
        int e = threadIdx.x + i * 128;
        int n = e >> 3, k4 = (e & 7) * 4;
        int ng = n0blk + n;
        float4 v = {0.f, 0.f, 0.f, 0.f};
        if (ng < N) v = *(const float4*)(Bb + (long)ng * sbn + (k0 + k4));
        v4bf pk = {f2bf(v.x), f2bf(v.y), f2bf(v.z), f2bf(v.w)};
        *(v4bf*)&lB[n * LBS + k4] = pk;
      }
    }
    __syncthreads();

    if (active) {
      if (k0 + 32 < K)
        __builtin_prefetch((const void*)(Bb + (sbn == 1 ? (long)(k0 + 32) * sbk + n0
                                                        : (long)n0 * sbn + (k0 + 32))), 0, 0);
      // A fragments (ISA 16-bit A layout: K = kh*8+j and 16+kh*8+j), via ds_load_b128
      v8bf a0l = *(const v8bf*)&lA[am * 32 + kh * 8];
      v8bf a0h = *(const v8bf*)&lA[am * 32 + 16 + kh * 8];
      v8bf a1l = *(const v8bf*)&lA[(am + 16) * 32 + kh * 8];
      v8bf a1h = *(const v8bf*)&lA[(am + 16) * 32 + 16 + kh * 8];
      v16bf a0 = __builtin_shufflevector(a0l, a0h, 0,1,2,3,4,5,6,7,8,9,10,11,12,13,14,15);
      v16bf a1 = __builtin_shufflevector(a1l, a1h, 0,1,2,3,4,5,6,7,8,9,10,11,12,13,14,15);
#pragma unroll
      for (int f = 0; f < 4; ++f) {
        const __bf16* bp = &lB[(wave * 64 + f * 16 + am) * LBS + kh * 8];
        v8bf b0 = *(const v8bf*)bp;
        v8bf b1 = *(const v8bf*)(bp + 16);
        v16bf bfv = __builtin_shufflevector(b0, b1, 0,1,2,3,4,5,6,7,8,9,10,11,12,13,14,15);
        acc[0][f] = __builtin_amdgcn_wmma_f32_16x16x32_bf16(
            false, a0, false, bfv, (short)0, acc[0][f], false, false);
        acc[1][f] = __builtin_amdgcn_wmma_f32_16x16x32_bf16(
            false, a1, false, bfv, (short)0, acc[1][f], false, false);
      }
    }
    __syncthreads();
  }

  if (active) {
    const int crow = kh * 8;
#pragma unroll
    for (int g = 0; g < 2; ++g) {
#pragma unroll
      for (int f = 0; f < 4; ++f) {
        const int n = n0 + f * 16 + am;
        const float bv = bias ? bias[n] : 0.0f;
#pragma unroll
        for (int r = 0; r < 8; ++r)
          Cb[(long)(m0 + g * 16 + r + crow) * ldc + n] = acc[g][f][r] + bv;
      }
    }
  }
}

// ---------------- embedding: gather + poincare proj + log0 + pos ----------------
__global__ __launch_bounds__(256)
void k_embed(const float* __restrict__ emb, const float* __restrict__ pos,
             const int* __restrict__ ids,
             float* __restrict__ hyp, float* __restrict__ eu,
             float* __restrict__ h, float* __restrict__ x2) {
  __shared__ float sh[256];
  const int tok = blockIdx.x;
  const int t = tok % TT;
  const long base = (long)ids[tok] * DM;
  float e[3]; float ss = 0.0f;
#pragma unroll
  for (int i = 0; i < 3; ++i) {
    int d = threadIdx.x + i * 256;
    e[i] = emb[base + d];
    ss += e[i] * e[i];
  }
  const float n2 = blk_sum(ss, sh);
  const float n = sqrtf(n2);
  const float maxnorm = 1.0f - 1e-4f;
  const float scale = (n > maxnorm) ? maxnorm / fmaxf(n, 1e-12f) : 1.0f;
  const float hn = fmaxf(n * scale, 1e-9f);
  const float fac = atanhf(fminf(hn, 1.0f - 1e-7f)) / hn;
  if (threadIdx.x == 0) x2[tok] = hn * hn;
#pragma unroll
  for (int i = 0; i < 3; ++i) {
    int d = threadIdx.x + i * 256;
    float hv = e[i] * scale;
    float ev = fac * hv;
    hyp[(long)tok * DM + d] = hv;
    eu[(long)tok * DM + d] = ev;
    h[(long)tok * DM + d] = ev + pos[(long)t * DM + d];
  }
}

// ---------------- pairwise capped poincare distance from xy, x2 ----------------
__global__ __launch_bounds__(256)
void k_dist(const float* __restrict__ xy, const float* __restrict__ x2,
            float* __restrict__ dd, int total) {
  int idx = blockIdx.x * blockDim.x + threadIdx.x;
  if (idx >= total) return;
  int j = idx % TT;
  int i = (idx / TT) % TT;
  int b = idx / (TT * TT);
  float xi = x2[b * TT + i], xj = x2[b * TT + j];
  float sq = fmaxf(xi + xj - 2.0f * xy[idx], 0.0f);
  float den = fmaxf((1.0f - xi) * (1.0f - xj), 1e-9f);
  float arg = fmaxf(1.0f + 2.0f * sq / den, 1.0f);
  dd[idx] = fminf(acoshf(arg), DMAXC);
}

// ---------------- struct scores: softmax over row, then alpha*log ----------------
__global__ __launch_bounds__(256)
void k_struct_softmax(float* __restrict__ sb) {
  __shared__ float sh[256];
  float* p = sb + (long)blockIdx.x * TT;
  float v[4]; float mx = -1e30f;
#pragma unroll
  for (int i = 0; i < 4; ++i) {
    v[i] = p[threadIdx.x + i * 256] * INV_SQRT_D;
    mx = fmaxf(mx, v[i]);
  }
  mx = blk_max(mx, sh);
  float s = 0.0f;
#pragma unroll
  for (int i = 0; i < 4; ++i) { v[i] = expf(v[i] - mx); s += v[i]; }
  s = blk_sum(s, sh);
  const float inv = 1.0f / s;
#pragma unroll
  for (int i = 0; i < 4; ++i)
    p[threadIdx.x + i * 256] = ALPHA_STRUCT * logf(v[i] * inv + 1e-8f);
}

// ---------------- attention softmax with biases (per b,h; grid = T rows) --------
__global__ __launch_bounds__(256)
void k_attn_softmax(float* __restrict__ sc, const float* __restrict__ sb,
                    const float* __restrict__ dd, const float* __restrict__ mask,
                    const float* __restrict__ braw, int head) {
  __shared__ float sh[256];
  const float b0 = braw[head];
  const float beta = (head < NH / 2) ? log1pf(expf(b0)) : 0.0f;
  float* p = sc + (long)blockIdx.x * TT;
  const float* sbr = sb + (long)blockIdx.x * TT;
  const float* ddr = dd + (long)blockIdx.x * TT;
  float v[4]; float mx = -1e30f;
#pragma unroll
  for (int i = 0; i < 4; ++i) {
    int j = threadIdx.x + i * 256;
    float val = p[j] * INV_SQRT_HD + sbr[j] - beta * ddr[j] + (1.0f - mask[j]) * (-1e9f);
    v[i] = val; mx = fmaxf(mx, val);
  }
  mx = blk_max(mx, sh);
  float s = 0.0f;
#pragma unroll
  for (int i = 0; i < 4; ++i) { v[i] = expf(v[i] - mx); s += v[i]; }
  s = blk_sum(s, sh);
  const float inv = 1.0f / s;
#pragma unroll
  for (int i = 0; i < 4; ++i) p[threadIdx.x + i * 256] = v[i] * inv;
}

// ---------------- (optional residual) + LayerNorm ----------------
__global__ __launch_bounds__(256)
void k_add_ln(const float* __restrict__ a, const float* __restrict__ r,
              const float* __restrict__ s, const float* __restrict__ bb,
              float* __restrict__ out) {
  __shared__ float sh[256];
  const long row = blockIdx.x;
  float x[3]; float sm = 0.0f, sq = 0.0f;
#pragma unroll
  for (int i = 0; i < 3; ++i) {
    int d = threadIdx.x + i * 256;
    float v = a[row * DM + d] + (r ? r[row * DM + d] : 0.0f);
    x[i] = v; sm += v; sq += v * v;
  }
  sm = blk_sum(sm, sh);
  sq = blk_sum(sq, sh);
  const float mean = sm * (1.0f / DM);
  const float var = sq * (1.0f / DM) - mean * mean;
  const float rstd = rsqrtf(fmaxf(var, 0.0f) + LN_EPS);
#pragma unroll
  for (int i = 0; i < 3; ++i) {
    int d = threadIdx.x + i * 256;
    out[row * DM + d] = (x[i] - mean) * rstd * s[d] + bb[d];
  }
}

// ---------------- tanh-approx GELU, float4 vectorized ----------------
__global__ __launch_bounds__(256)
void k_gelu4(float4* __restrict__ x, long n4) {
  long i = (long)blockIdx.x * blockDim.x + threadIdx.x;
  if (i < n4) {
    float4 v = x[i];
    float vv[4] = {v.x, v.y, v.z, v.w};
#pragma unroll
    for (int j = 0; j < 4; ++j) {
      float c = vv[j] + 0.044715f * vv[j] * vv[j] * vv[j];
      vv[j] = 0.5f * vv[j] * (1.0f + tanhf(0.7978845608028654f * c));
    }
    x[i] = {vv[0], vv[1], vv[2], vv[3]};
  }
}

// ---------------- host-side GEMM launcher ----------------
static inline void gemm(hipStream_t st,
                        const float* A, long lda, long sAz,
                        const float* B, long sbk, long sbn, long sBz,
                        float* C, long ldc, long sCz, const float* bias,
                        int M, int N, int K, int Z) {
  dim3 g(M / 32, (N + 255) / 256, Z);
  wmma_gemm<<<g, 128, 0, st>>>(A, lda, sAz, B, sbk, sbn, sBz, C, ldc, sCz, bias, M, N, K);
}

extern "C" void kernel_launch(void* const* d_in, const int* in_sizes, int n_in,
                              void* d_out, int out_size, void* d_ws, size_t ws_size,
                              hipStream_t stream) {
  (void)in_sizes; (void)n_in; (void)out_size; (void)ws_size;
  // ---- inputs, setup_inputs() dict insertion order ----
  const float* emb    = (const float*)d_in[0];   // [V,D]
  const float* pos    = (const float*)d_in[1];   // [L,D]
  const float* lnf_s  = (const float*)d_in[2];
  const float* lnf_b  = (const float*)d_in[3];
  const float* Wout   = (const float*)d_in[4];   // [D,V]
  const float* Wq     = (const float*)d_in[5];   // [NL,D,D]
  const float* Wk     = (const float*)d_in[6];
  const float* Wv     = (const float*)d_in[7];
  const float* Wo     = (const float*)d_in[8];
  const float* bo     = (const float*)d_in[9];   // [NL,D]
  const float* Wsh    = (const float*)d_in[10];
  const float* bsh    = (const float*)d_in[11];
  const float* Wsd    = (const float*)d_in[12];
  const float* bsd    = (const float*)d_in[13];
  const float* braw   = (const float*)d_in[14];  // [NL,H]
  const float* ln1_s  = (const float*)d_in[15];
  const float* ln1_b  = (const float*)d_in[16];
  const float* W1     = (const float*)d_in[17];  // [NL,D,4D]
  const float* b1     = (const float*)d_in[18];  // [NL,4D]
  const float* W2     = (const float*)d_in[19];  // [NL,4D,D]
  const float* b2     = (const float*)d_in[20];
  const float* ln2_s  = (const float*)d_in[21];
  const float* ln2_b  = (const float*)d_in[22];
  const int*   ids    = (const int*)d_in[23];    // [B,T]
  const float* amask  = (const float*)d_in[24];  // [B,T]

  // ---- outputs: logits | hyp | eu (fp32, concatenated) ----
  float* out    = (float*)d_out;
  float* logits = out;
  float* hypo   = out + (size_t)BB * TT * VSZ;
  float* euo    = hypo + (size_t)BB * TT * DM;

  // ---- workspace partition ----
  float* w = (float*)d_ws;
  size_t off = 0;
  auto alloc = [&](size_t n) { float* p = w + off; off += (n + 63) & ~(size_t)63; return p; };
  float* h     = alloc((size_t)MM * DM);
  float* h1    = alloc((size_t)MM * DM);
  float* t0    = alloc((size_t)MM * DM);
  float* headb = alloc((size_t)MM * DM);
  float* depb  = alloc((size_t)MM * DM);
  float* qb    = alloc((size_t)MM * DM);
  float* kb    = alloc((size_t)MM * DM);
  float* vb    = alloc((size_t)MM * DM);
  float* ctxb  = alloc((size_t)MM * DM);
  float* f1    = alloc((size_t)MM * 4 * DM);
  float* sbias = alloc((size_t)BB * TT * TT);
  float* dist  = alloc((size_t)BB * TT * TT);
  float* xy    = alloc((size_t)BB * TT * TT);
  float* x2    = alloc((size_t)MM);
  float* sc    = alloc((size_t)TT * TT);

  // 1) embedding: hyp, eu, h = eu + pos, x2
  k_embed<<<MM, 256, 0, stream>>>(emb, pos, ids, hypo, euo, h, x2);

  // 2) xy = hyp @ hyp^T per batch (WMMA, z over B), then distances
  gemm(stream, hypo, DM, (long)TT * DM,
       hypo, 1, DM, (long)TT * DM,
       xy, TT, (long)TT * TT, nullptr, TT, TT, DM, BB);
  {
    int total = BB * TT * TT;
    k_dist<<<(total + 255) / 256, 256, 0, stream>>>(xy, x2, dist, total);
  }

  // 3) layers
  for (int l = 0; l < NLAY; ++l) {
    const float* Wq_l = Wq + (size_t)l * DM * DM;
    const float* Wk_l = Wk + (size_t)l * DM * DM;
    const float* Wv_l = Wv + (size_t)l * DM * DM;
    const float* Wo_l = Wo + (size_t)l * DM * DM;
    const float* Wsh_l = Wsh + (size_t)l * DM * DM;
    const float* Wsd_l = Wsd + (size_t)l * DM * DM;
    const float* W1_l = W1 + (size_t)l * DM * 4 * DM;
    const float* W2_l = W2 + (size_t)l * 4 * DM * DM;
    const float* bo_l = bo + (size_t)l * DM;
    const float* bsh_l = bsh + (size_t)l * DM;
    const float* bsd_l = bsd + (size_t)l * DM;
    const float* b1_l = b1 + (size_t)l * 4 * DM;
    const float* b2_l = b2 + (size_t)l * DM;
    const float* braw_l = braw + (size_t)l * NH;

    // soft-parsing head/dep projections
    gemm(stream, h, DM, 0, Wsh_l, DM, 1, 0, headb, DM, 0, bsh_l, MM, DM, DM, 1);
    gemm(stream, h, DM, 0, Wsd_l, DM, 1, 0, depb, DM, 0, bsd_l, MM, DM, DM, 1);
    // ss_raw[b] = head_b @ dep_b^T   (z over batch)
    gemm(stream, headb, DM, (long)TT * DM,
         depb, 1, DM, (long)TT * DM,
         sbias, TT, (long)TT * TT, nullptr, TT, TT, DM, BB);
    k_struct_softmax<<<MM, 256, 0, stream>>>(sbias);

    // q, k, v projections (no bias)
    gemm(stream, h, DM, 0, Wq_l, DM, 1, 0, qb, DM, 0, nullptr, MM, DM, DM, 1);
    gemm(stream, h, DM, 0, Wk_l, DM, 1, 0, kb, DM, 0, nullptr, MM, DM, DM, 1);
    gemm(stream, h, DM, 0, Wv_l, DM, 1, 0, vb, DM, 0, nullptr, MM, DM, DM, 1);

    // attention per (batch, head): scores -> softmax(+biases) -> ctx
    for (int b = 0; b < BB; ++b) {
      for (int hh = 0; hh < NH; ++hh) {
        const long bo_off = (long)b * TT * DM + (long)hh * HD;
        gemm(stream, qb + bo_off, DM, 0,
             kb + bo_off, 1, DM, 0,
             sc, TT, 0, nullptr, TT, TT, HD, 1);
        k_attn_softmax<<<TT, 256, 0, stream>>>(
            sc, sbias + (long)b * TT * TT, dist + (long)b * TT * TT,
            amask + (long)b * TT, braw_l, hh);
        gemm(stream, sc, TT, 0,
             vb + bo_off, DM, 1, 0,
             ctxb + bo_off, DM, 0, nullptr, TT, HD, TT, 1);
      }
    }

    // output projection + residual + LN1
    gemm(stream, ctxb, DM, 0, Wo_l, DM, 1, 0, t0, DM, 0, bo_l, MM, DM, DM, 1);
    k_add_ln<<<MM, 256, 0, stream>>>(h, t0, ln1_s + (size_t)l * DM, ln1_b + (size_t)l * DM, h1);

    // FFN
    gemm(stream, h1, DM, 0, W1_l, 4 * DM, 1, 0, f1, 4 * DM, 0, b1_l, MM, 4 * DM, DM, 1);
    {
      long n4 = (long)MM * 4 * DM / 4;
      k_gelu4<<<(unsigned)((n4 + 255) / 256), 256, 0, stream>>>((float4*)f1, n4);
    }
    gemm(stream, f1, 4 * DM, 0, W2_l, DM, 1, 0, t0, DM, 0, b2_l, MM, DM, 4 * DM, 1);
    k_add_ln<<<MM, 256, 0, stream>>>(h1, t0, ln2_s + (size_t)l * DM, ln2_b + (size_t)l * DM, h);
  }

  // 4) final LN + logits head (768 -> 32000)
  k_add_ln<<<MM, 256, 0, stream>>>(h, nullptr, lnf_s, lnf_b, h1);
  gemm(stream, h1, DM, 0, Wout, VSZ, 1, 0, logits, VSZ, 0, nullptr, MM, VSZ, DM, 1);
}
